// TalkerForExport_66864050864548
// MI455X (gfx1250) — matrix operand order repaired and checked by reading, not compiled, not run on hardware
//
#include <hip/hip_runtime.h>
#include <hip/hip_bf16.h>
#include <cstdint>
#include <cstddef>

// Model dims (fixed by the reference)
#define L_  12
#define D_  1024
#define H_  16
#define KV_ 8
#define HD_ 64
#define F_  4096
#define V_  8192
#define T_  1024

typedef _Float16 v8h  __attribute__((ext_vector_type(8)));
typedef _Float16 v16h __attribute__((ext_vector_type(16)));
typedef float    v8f  __attribute__((ext_vector_type(8)));

// ---------------------------------------------------------------------------
// GEMM: C[M,N] = A[M,K] @ B[K,N] (+bias[N]) (+addsrc[M,N])
// fp32 in -> f16 tiles in LDS -> v_wmma_f32_16x16x32_f16 -> fp32 out.
// Block: 128 threads = 4 waves (2x2), block tile 64x128, wave tile 32x64.
// ---------------------------------------------------------------------------
#define BM 64
#define BN 128
#define BK 32
#define SA 40   // LDS row stride (f16 elems) for A tile [64][32] (pad vs banks)
#define SB 48   // LDS row stride (f16 elems) for B tile stored as [N=128][K=32]

__global__ __launch_bounds__(128) void gemm_f16w_kernel(
    const float* __restrict__ A, const float* __restrict__ B,
    const float* __restrict__ bias, const float* __restrict__ addsrc,
    float* __restrict__ C, int M, int N, int K)
{
  __shared__ __align__(16) _Float16 la[BM * SA];
  __shared__ __align__(32) _Float16 lb[BN * SB];

  const int tid  = threadIdx.x;
  const int wave = tid >> 5;
  const int lane = tid & 31;
  const int m0   = blockIdx.y * BM;
  const int n0   = blockIdx.x * BN;
  const int wrow = wave >> 1;   // 0..1 -> 32-row strip
  const int wcol = wave & 1;    // 0..1 -> 64-col strip

  v8f acc[2][4];
#pragma unroll
  for (int i = 0; i < 2; ++i)
#pragma unroll
    for (int j = 0; j < 4; ++j)
      acc[i][j] = v8f{0.f, 0.f, 0.f, 0.f, 0.f, 0.f, 0.f, 0.f};

  // Per-lane fragment addressing (contiguous k order, same for A and B):
  // A 16x32: lane<16 holds row=lane, K 0..15; lane>=16 row=lane-16, K 16..31
  // B 32x16 (stored [n][k] in LDS): lane<16 col=lane K 0..15; lane>=16 col=lane-16 K 16..31
  const int arow = lane & 15;
  const int klo  = (lane < 16) ? 0 : 16;
  const int kb   = (lane < 16) ? 0 : 16;
  const int col  = lane & 15;

  for (int k0 = 0; k0 < K; k0 += BK) {
    // ---- Stage A tile 64x32 (row-major, cvt f32->f16). 2 threads per row.
    {
      const int row  = tid >> 1;
      const int half = (tid & 1) << 4;
      const float* src = A + (size_t)(m0 + row) * K + k0 + half;
      _Float16* dst = &la[row * SA + half];
#pragma unroll
      for (int i = 0; i < 16; i += 4) {
        float4 f = *(const float4*)(src + i);
        dst[i + 0] = (_Float16)f.x;
        dst[i + 1] = (_Float16)f.y;
        dst[i + 2] = (_Float16)f.z;
        dst[i + 3] = (_Float16)f.w;
      }
    }
    // ---- Stage B tile 32x128 transposed into lb[n][k] (coalesced global reads)
    {
      const float* src = B + (size_t)k0 * N + n0 + tid;
      _Float16* dst = &lb[tid * SB];
#pragma unroll 4
      for (int kk = 0; kk < BK; ++kk)
        dst[kk] = (_Float16)src[(size_t)kk * N];
    }
    if (k0 + BK < K)
      __builtin_prefetch(B + (size_t)(k0 + BK) * N + n0 + tid, 0, 1);
    __syncthreads();

    // ---- Fragments: load everything first, then the WMMA burst
    v16h afrag[2];
#pragma unroll
    for (int i = 0; i < 2; ++i) {
      const _Float16* ap = &la[(wrow * 32 + i * 16 + arow) * SA + klo];
      v8h lo = *(const v8h*)ap;
      v8h hi = *(const v8h*)(ap + 8);
      afrag[i] = __builtin_shufflevector(lo, hi, 0, 1, 2, 3, 4, 5, 6, 7,
                                         8, 9, 10, 11, 12, 13, 14, 15);
    }
    v16h bfrag[4];
#pragma unroll
    for (int j = 0; j < 4; ++j)
      bfrag[j] = *(const v16h*)&lb[(wcol * 64 + j * 16 + col) * SB + kb];
#pragma unroll
    for (int j = 0; j < 4; ++j)
#pragma unroll
      for (int i = 0; i < 2; ++i)
        acc[i][j] = __builtin_amdgcn_wmma_f32_16x16x32_f16(
            false, afrag[i], false, bfrag[j], (short)0, acc[i][j], false, false);
    __syncthreads();
  }

  // ---- Epilogue. C-layout: VGPR r -> rows r (lanes 0-15) / r+8 (lanes 16-31)
  const int lrow = (lane < 16) ? 0 : 8;
#pragma unroll
  for (int i = 0; i < 2; ++i)
#pragma unroll
    for (int j = 0; j < 4; ++j)
#pragma unroll
      for (int r = 0; r < 8; ++r) {
        const int m = m0 + wrow * 32 + i * 16 + r + lrow;
        const int n = n0 + wcol * 64 + j * 16 + col;
        float v = acc[i][j][r];
        if (bias)   v += bias[n];
        if (addsrc) v += addsrc[(size_t)m * N + n];
        C[(size_t)m * N + n] = v;
      }
}

// ---------------------------------------------------------------------------
// RMSNorm: out[row] = h[row] * rsqrt(mean(h^2)+eps) * w
// ---------------------------------------------------------------------------
__global__ __launch_bounds__(256) void rmsnorm_kernel(
    const float* __restrict__ h, const float* __restrict__ w,
    float* __restrict__ out)
{
  const int row = blockIdx.x;
  const float* x = h + (size_t)row * D_;
  float s = 0.f;
  for (int i = threadIdx.x; i < D_; i += 256) {
    float v = x[i];
    s += v * v;
  }
#pragma unroll
  for (int off = 16; off > 0; off >>= 1) s += __shfl_down(s, off, 32);
  __shared__ float red[8];
  if ((threadIdx.x & 31) == 0) red[threadIdx.x >> 5] = s;
  __syncthreads();
  if (threadIdx.x == 0) {
    float t = 0.f;
#pragma unroll
    for (int i = 0; i < 8; ++i) t += red[i];
    red[0] = rsqrtf(t / (float)D_ + 1e-6f);
  }
  __syncthreads();
  const float r = red[0];
  for (int i = threadIdx.x; i < D_; i += 256)
    out[(size_t)row * D_ + i] = x[i] * r * w[i];
}

// ---------------------------------------------------------------------------
// RoPE in-place on [T, nheads*64]: pair (d, d+32), theta = 1e6
// ---------------------------------------------------------------------------
__global__ __launch_bounds__(256) void rope_kernel(
    float* __restrict__ q, const int* __restrict__ pos_ids, int nheads)
{
  const int idx = blockIdx.x * 256 + threadIdx.x;
  const int i = idx & 31;                 // freq index 0..31
  const int h = (idx >> 5) % nheads;
  const int t = idx / (32 * nheads);
  if (t >= T_) return;
  const float pos = (float)pos_ids[t];
  const float inv = __powf(1.0e6f, -(2.0f * (float)i) / 64.0f);
  const float ang = pos * inv;
  const float c = __cosf(ang), s = __sinf(ang);
  float* p = q + (size_t)t * nheads * HD_ + h * HD_;
  const float x1 = p[i];
  const float x2 = p[i + 32];
  p[i]      = x1 * c - x2 * s;
  p[i + 32] = x2 * c + x1 * s;
}

// ---------------------------------------------------------------------------
// Flash attention, WMMA version. One wave32 per (16-query tile, head).
// q/attn: [T, H*64]; k/v: [T, KV*64]; head h uses kv-head h/2 (GQA).
// Per 32-key chunk: S = Q@K^T via 4 WMMAs, causal mask + online softmax on
// the C-fragment layout (row r / r+8 lives in a 16-lane half-group), P goes
// through LDS to be reshaped C-layout -> A-layout, then O += P@V (4 WMMAs).
// Single-wave workgroup: barrier ops degrade to S_NOP + waits (safe in-loop).
// ---------------------------------------------------------------------------
#define SKV 80   // LDS stride for K chunk [key=32][hd=64] (160B, 32B-aligned)
#define SVT 48   // LDS stride for V^T chunk [hd=64][key=32]
#define SP  40   // LDS stride for P tile [row=16][key=32]

__global__ __launch_bounds__(32) void attn_wmma_kernel(
    const float* __restrict__ q, const float* __restrict__ k,
    const float* __restrict__ v, float* __restrict__ out)
{
  __shared__ __align__(32) _Float16 lk[32 * SKV];  // K chunk; also Q staging
  __shared__ __align__(32) _Float16 lv[64 * SVT];  // V^T chunk
  __shared__ __align__(16) _Float16 lp[16 * SP];   // P tile

  const int lane = threadIdx.x;
  const int q0   = blockIdx.x * 16;
  const int h    = blockIdx.y;
  const int kvh  = h >> 1;                  // H/KV = 2
  const int g    = (lane < 16) ? 0 : 1;     // half-group
  const int c15  = lane & 15;
  const int klo  = g ? 16 : 0;              // A-fragment K base
  const int kb   = g ? 16 : 0;              // B-fragment K base

  // ---- Stage Q tile 16x64 into lk, then build the two Q A-fragments.
  {
    const int row  = lane & 15;
    const int half = (lane >> 4) * 32;
    const float* src = q + (size_t)(q0 + row) * (H_ * HD_) + h * HD_ + half;
    _Float16* dst = &lk[row * SKV + half];
#pragma unroll
    for (int i = 0; i < 32; i += 4) {
      float4 f = *(const float4*)(src + i);
      dst[i + 0] = (_Float16)f.x;
      dst[i + 1] = (_Float16)f.y;
      dst[i + 2] = (_Float16)f.z;
      dst[i + 3] = (_Float16)f.w;
    }
  }
  __syncthreads();
  v16h qfrag[2];
#pragma unroll
  for (int c = 0; c < 2; ++c) {
    const _Float16* ap = &lk[c15 * SKV + c * 32 + klo];
    v8h lo = *(const v8h*)ap;
    v8h hi = *(const v8h*)(ap + 8);
    qfrag[c] = __builtin_shufflevector(lo, hi, 0, 1, 2, 3, 4, 5, 6, 7,
                                       8, 9, 10, 11, 12, 13, 14, 15);
  }
  __syncthreads();

  v8f o[4];
#pragma unroll
  for (int j = 0; j < 4; ++j)
    o[j] = v8f{0.f, 0.f, 0.f, 0.f, 0.f, 0.f, 0.f, 0.f};
  float mrow[8], lrow[8];
#pragma unroll
  for (int r = 0; r < 8; ++r) { mrow[r] = -3.0e38f; lrow[r] = 0.f; }

  const int nchunk = (q0 + 16 + 31) / 32;   // causal: keys < q0+16
  for (int ch = 0; ch < nchunk; ++ch) {
    const int kc = ch * 32;
    // ---- Stage K chunk [key][hd] (B-fragment-friendly: contiguous hd)
    {
      const float* src = k + (size_t)(kc + lane) * (KV_ * HD_) + kvh * HD_;
      _Float16* dst = &lk[lane * SKV];
#pragma unroll
      for (int i = 0; i < 64; i += 4) {
        float4 f = *(const float4*)(src + i);
        dst[i + 0] = (_Float16)f.x;
        dst[i + 1] = (_Float16)f.y;
        dst[i + 2] = (_Float16)f.z;
        dst[i + 3] = (_Float16)f.w;
      }
    }
    // ---- Stage V^T chunk [hd][key] (transpose so keys are contiguous)
    {
      const float* src = v + (size_t)(kc + lane) * (KV_ * HD_) + kvh * HD_;
#pragma unroll 8
      for (int i = 0; i < 64; ++i)
        lv[i * SVT + lane] = (_Float16)src[i];
    }
    __syncthreads();

    // ---- Scores: two 16x16 tiles, contraction over hd (2 steps of 32)
    v8f s[2];
#pragma unroll
    for (int n = 0; n < 2; ++n) {
      s[n] = v8f{0.f, 0.f, 0.f, 0.f, 0.f, 0.f, 0.f, 0.f};
#pragma unroll
      for (int c = 0; c < 2; ++c) {
        v16h bfrag = *(const v16h*)&lk[(n * 16 + c15) * SKV + c * 32 + kb];
        s[n] = __builtin_amdgcn_wmma_f32_16x16x32_f16(
            false, qfrag[c], false, bfrag, (short)0, s[n], false, false);
      }
    }
    // ---- Scale + causal mask (selects, no branches: EXEC stays all-ones)
#pragma unroll
    for (int n = 0; n < 2; ++n)
#pragma unroll
      for (int r = 0; r < 8; ++r) {
        const int row  = q0 + r + 8 * g;
        const int colk = kc + n * 16 + c15;
        const float sv = s[n][r] * 0.125f;
        s[n][r] = (colk <= row) ? sv : -3.0e38f;
      }
    // ---- Online softmax on C-layout: reduce across 16-lane half-groups
#pragma unroll
    for (int r = 0; r < 8; ++r) {
      float mx = fmaxf(s[0][r], s[1][r]);
#pragma unroll
      for (int off = 1; off < 16; off <<= 1)
        mx = fmaxf(mx, __shfl_xor(mx, off, 32));
      const float mnew = fmaxf(mrow[r], mx);
      const float corr = __expf(mrow[r] - mnew);
      mrow[r] = mnew;
      const float p0 = __expf(s[0][r] - mnew);
      const float p1 = __expf(s[1][r] - mnew);
      s[0][r] = p0;
      s[1][r] = p1;
      float ps = p0 + p1;
#pragma unroll
      for (int off = 1; off < 16; off <<= 1)
        ps += __shfl_xor(ps, off, 32);
      lrow[r] = lrow[r] * corr + ps;
#pragma unroll
      for (int j = 0; j < 4; ++j) o[j][r] *= corr;
    }
    // ---- P: C-layout -> LDS -> A-layout fragment
#pragma unroll
    for (int r = 0; r < 8; ++r) {
      lp[(r + 8 * g) * SP + c15]      = (_Float16)s[0][r];
      lp[(r + 8 * g) * SP + 16 + c15] = (_Float16)s[1][r];
    }
    __syncthreads();
    v16h pfrag;
    {
      const _Float16* ap = &lp[c15 * SP + klo];
      v8h lo = *(const v8h*)ap;
      v8h hi = *(const v8h*)(ap + 8);
      pfrag = __builtin_shufflevector(lo, hi, 0, 1, 2, 3, 4, 5, 6, 7,
                                      8, 9, 10, 11, 12, 13, 14, 15);
    }
    // ---- O += P @ V (4 hd-column tiles)
#pragma unroll
    for (int j = 0; j < 4; ++j) {
      v16h vfrag = *(const v16h*)&lv[(j * 16 + c15) * SVT + kb];
      o[j] = __builtin_amdgcn_wmma_f32_16x16x32_f16(
          false, pfrag, false, vfrag, (short)0, o[j], false, false);
    }
    __syncthreads();
  }

  // ---- Epilogue: O /= rowsum, write [T, H*64]
#pragma unroll
  for (int j = 0; j < 4; ++j)
#pragma unroll
    for (int r = 0; r < 8; ++r) {
      const int row = q0 + r + 8 * g;
      const int col = h * HD_ + j * 16 + c15;
      out[(size_t)row * (H_ * HD_) + col] = o[j][r] / lrow[r];
    }
}

// ---------------------------------------------------------------------------
// g = silu(g) * u
// ---------------------------------------------------------------------------
__global__ __launch_bounds__(256) void silu_mul_kernel(
    float* __restrict__ g, const float* __restrict__ u, int n)
{
  const int i = blockIdx.x * 256 + threadIdx.x;
  if (i < n) {
    const float x = g[i];
    g[i] = (x / (1.f + __expf(-x))) * u[i];
  }
}

// ---------------------------------------------------------------------------
extern "C" void kernel_launch(void* const* d_in, const int* in_sizes, int n_in,
                              void* d_out, int out_size, void* d_ws, size_t ws_size,
                              hipStream_t stream)
{
  (void)in_sizes; (void)n_in; (void)out_size; (void)ws_size;
  const float* inputs = (const float*)d_in[0];
  const int*   pos    = (const int*)  d_in[1];   // (3,1,T) int32; use first copy
  const float* ln1    = (const float*)d_in[2];
  const float* wq     = (const float*)d_in[3];
  const float* bq     = (const float*)d_in[4];
  const float* wk     = (const float*)d_in[5];
  const float* bk     = (const float*)d_in[6];
  const float* wv     = (const float*)d_in[7];
  const float* bv     = (const float*)d_in[8];
  const float* wo     = (const float*)d_in[9];
  const float* ln2    = (const float*)d_in[10];
  const float* wg     = (const float*)d_in[11];
  const float* wu     = (const float*)d_in[12];
  const float* wd     = (const float*)d_in[13];
  const float* normw  = (const float*)d_in[14];
  const float* headw  = (const float*)d_in[15];
  float* out = (float*)d_out;

  // Workspace layout (fp32): 13M floats = 52 MB
  float* ws   = (float*)d_ws;
  float* h    = ws;                                 // T*D
  float* x    = h    + (size_t)T_ * D_;             // T*D
  float* q    = x    + (size_t)T_ * D_;             // T*H*HD = T*D
  float* kbuf = q    + (size_t)T_ * D_;             // T*KV*HD
  float* vbuf = kbuf + (size_t)T_ * KV_ * HD_;      // T*KV*HD
  float* attn = vbuf + (size_t)T_ * KV_ * HD_;      // T*D
  float* g    = attn + (size_t)T_ * D_;             // T*F
  float* u    = g    + (size_t)T_ * F_;             // T*F

  hipMemcpyAsync(h, inputs, (size_t)T_ * D_ * sizeof(float),
                 hipMemcpyDeviceToDevice, stream);

  const dim3 gD (D_ / BN, T_ / BM);     // N=1024 GEMMs
  const dim3 gKV((KV_ * HD_) / BN, T_ / BM);
  const dim3 gF (F_ / BN, T_ / BM);
  const dim3 gV (V_ / BN, T_ / BM);

  for (int l = 0; l < L_; ++l) {
    const float* p_ln1 = ln1 + (size_t)l * D_;
    const float* p_wq  = wq  + (size_t)l * D_ * (H_ * HD_);
    const float* p_bq  = bq  + (size_t)l * (H_ * HD_);
    const float* p_wk  = wk  + (size_t)l * D_ * (KV_ * HD_);
    const float* p_bk  = bk  + (size_t)l * (KV_ * HD_);
    const float* p_wv  = wv  + (size_t)l * D_ * (KV_ * HD_);
    const float* p_bv  = bv  + (size_t)l * (KV_ * HD_);
    const float* p_wo  = wo  + (size_t)l * (H_ * HD_) * D_;
    const float* p_ln2 = ln2 + (size_t)l * D_;
    const float* p_wg  = wg  + (size_t)l * D_ * F_;
    const float* p_wu  = wu  + (size_t)l * D_ * F_;
    const float* p_wd  = wd  + (size_t)l * F_ * D_;

    rmsnorm_kernel<<<T_, 256, 0, stream>>>(h, p_ln1, x);
    gemm_f16w_kernel<<<gD,  128, 0, stream>>>(x, p_wq, p_bq, nullptr, q,    T_, H_ * HD_,  D_);
    gemm_f16w_kernel<<<gKV, 128, 0, stream>>>(x, p_wk, p_bk, nullptr, kbuf, T_, KV_ * HD_, D_);
    gemm_f16w_kernel<<<gKV, 128, 0, stream>>>(x, p_wv, p_bv, nullptr, vbuf, T_, KV_ * HD_, D_);
    rope_kernel<<<(T_ * H_  * 32) / 256, 256, 0, stream>>>(q,    pos, H_);
    rope_kernel<<<(T_ * KV_ * 32) / 256, 256, 0, stream>>>(kbuf, pos, KV_);
    attn_wmma_kernel<<<dim3(T_ / 16, H_), 32, 0, stream>>>(q, kbuf, vbuf, attn);
    gemm_f16w_kernel<<<gD, 128, 0, stream>>>(attn, p_wo, nullptr, h, h, T_, D_, H_ * HD_);
    rmsnorm_kernel<<<T_, 256, 0, stream>>>(h, p_ln2, x);
    gemm_f16w_kernel<<<gF, 128, 0, stream>>>(x, p_wg, nullptr, nullptr, g, T_, F_, D_);
    gemm_f16w_kernel<<<gF, 128, 0, stream>>>(x, p_wu, nullptr, nullptr, u, T_, F_, D_);
    silu_mul_kernel<<<(T_ * F_) / 256, 256, 0, stream>>>(g, u, T_ * F_);
    gemm_f16w_kernel<<<gD, 128, 0, stream>>>(g, p_wd, nullptr, h, h, T_, D_, F_);
  }

  // Final norm -> hidden output, then LM head -> logits output.
  float* hid_out = out + (size_t)T_ * V_;           // tuple: (logits, hidden)
  rmsnorm_kernel<<<T_, 256, 0, stream>>>(h, normw, hid_out);
  gemm_f16w_kernel<<<gV, 128, 0, stream>>>(hid_out, headw, nullptr, nullptr, out, T_, V_, D_);
}
